// TransformerEncoderLayer_8461085573377
// MI455X (gfx1250) — matrix operand order repaired and checked by reading, not compiled, not run on hardware
//
#include <hip/hip_runtime.h>
#include <hip/hip_bf16.h>
#include <math.h>

typedef __attribute__((ext_vector_type(16))) __bf16 v16bf;
typedef __attribute__((ext_vector_type(8)))  __bf16 v8bf;
typedef __attribute__((ext_vector_type(8)))  float  v8f;
typedef __attribute__((ext_vector_type(4)))  float  v4f;
typedef __attribute__((ext_vector_type(4)))  int    v4i;

#define TEL_D     1024
#define TEL_F     4096
#define TEL_S     2048
#define TEL_B     2
#define TEL_HD    64

// ---------------------------------------------------------------------------
// Async global->LDS staging (CDNA5 GLOBAL_LOAD_ASYNC_TO_LDS_B128, ASYNCcnt).
// Probe round confirmed the builtin exists and its param 0 is a non-const
// global-AS pointer to a 16B int vector.
// ---------------------------------------------------------------------------
#if __has_builtin(__builtin_amdgcn_global_load_async_to_lds_b128)
#define TEL_ASYNC 1
#else
#define TEL_ASYNC 0
#endif

__device__ __forceinline__ void stage16(const float* __restrict__ g, float* l) {
#if TEL_ASYNC
    __builtin_amdgcn_global_load_async_to_lds_b128(
        (__attribute__((address_space(1))) v4i*)g,
        (__attribute__((address_space(3))) v4i*)l,
        0, 0);
#else
    *(v4f*)l = *(const v4f*)g;
#endif
}

__device__ __forceinline__ void stage_wait() {
#if TEL_ASYNC
#if __has_builtin(__builtin_amdgcn_s_wait_asynccnt)
    __builtin_amdgcn_s_wait_asynccnt(0);
#else
    asm volatile("s_wait_asynccnt 0" ::: "memory");
#endif
#endif
}

// LDS row strides (elements). Chosen so 16-lane fragment reads hit 16 distinct
// bank groups (byte stride/4 has order 16 mod 64) and rows stay 16B-aligned.
#define AS_STRIDE 36   // fp32  rows: 144 B
#define BS_STRIDE 56   // bf16  rows: 112 B
#define KS_STRIDE 72   // bf16  rows: 144 B

// Build v16bf A-fragment from one fp32 LDS row (K runs {hi*8.., 16+hi*8..}).
__device__ __forceinline__ v16bf afrag_from_f32(const float* row, int hi) {
    const v4f* p0 = (const v4f*)(row + hi * 8);        // K = hi*8 .. +7
    const v4f* p1 = (const v4f*)(row + 16 + hi * 8);   // K = 16+hi*8 .. +7
    v4f f0 = p0[0], f1 = p0[1], f2 = p1[0], f3 = p1[1];
    v16bf r;
#pragma unroll
    for (int e = 0; e < 4; ++e) {
        r[e]      = (__bf16)f0[e];
        r[4 + e]  = (__bf16)f1[e];
        r[8 + e]  = (__bf16)f2[e];
        r[12 + e] = (__bf16)f3[e];
    }
    return r;
}

// Build v16bf B-fragment from a bf16 LDS row (ptr at hi*8): K runs {hi*8.., 16+hi*8..}.
__device__ __forceinline__ v16bf bfrag_from_bf16(const __bf16* row_k16) {
    v8bf a = *(const v8bf*)(row_k16);
    v8bf b = *(const v8bf*)(row_k16 + 16);
    return __builtin_shufflevector(a, b, 0,1,2,3,4,5,6,7,8,9,10,11,12,13,14,15);
}

// ---------------------------------------------------------------------------
// Generic bf16-WMMA GEMM: C[M,N] = act(A[M,K] @ B[K,N] + bias[N])
// BM=128, BN=128, BK=32. 256 threads = 8 waves (4 M x 2 N); each wave does
// 32x64 = 2x4 v_wmma_f32_16x16x32_bf16 per K-step.
// ---------------------------------------------------------------------------
__global__ __launch_bounds__(256) void gemm_bf16_wmma(
    const float* __restrict__ A, const float* __restrict__ B,
    const float* __restrict__ bias, float* __restrict__ C,
    int M, int N, int K, int relu)
{
    __shared__ float  Asf[128 * AS_STRIDE];   // 18 KB, fp32 row-major [r][k]
    __shared__ __bf16 Bst[128 * BS_STRIDE];   // 14 KB, bf16 TRANSPOSED [n][k]

    const int tid  = threadIdx.x;
    const int lane = tid & 31;
    const int wave = tid >> 5;
    const int waveM = wave >> 1;          // 0..3
    const int waveN = wave & 1;           // 0..1
    const int lo = lane & 15;
    const int hi = lane >> 4;
    const int rowBase = blockIdx.y * 128;
    const int colBase = blockIdx.x * 128;

    const v8f vz = {0.f,0.f,0.f,0.f,0.f,0.f,0.f,0.f};
    v8f acc[2][4];
#pragma unroll
    for (int tm = 0; tm < 2; ++tm)
#pragma unroll
        for (int tn = 0; tn < 4; ++tn) acc[tm][tn] = vz;

    // staging coordinates (fixed per thread)
    const int a_r  = tid >> 1;            // 0..127 (2 threads per row)
    const int a_c4 = (tid & 1) * 4;       // chunk columns a_c4 + {0,8,16,24}
    const int b_kk = tid >> 4;            // 0..15 -> two k rows per thread
    const int b_n8 = (tid & 15) * 8;      // 8 consecutive n

    for (int k0 = 0; k0 < K; k0 += 32) {
        // ---- stage A tile 128x32 fp32 (async path: no VGPR round-trip) ----
#pragma unroll
        for (int i = 0; i < 4; ++i) {
            int c = a_c4 + i * 8;
            stage16(&A[(size_t)(rowBase + a_r) * K + k0 + c],
                    &Asf[a_r * AS_STRIDE + c]);
        }
        // ---- stage B tile 32x128 -> bf16 transposed [n][k] ----
#pragma unroll
        for (int half = 0; half < 2; ++half) {
            int kk = b_kk + half * 16;
            const v4f* gp = (const v4f*)&B[(size_t)(k0 + kk) * N + colBase + b_n8];
            v4f f0 = gp[0], f1 = gp[1];
#pragma unroll
            for (int j = 0; j < 4; ++j) {
                Bst[(b_n8 + j) * BS_STRIDE + kk]     = (__bf16)f0[j];
                Bst[(b_n8 + 4 + j) * BS_STRIDE + kk] = (__bf16)f1[j];
            }
        }
        if (k0 + 32 < K) {
            __builtin_prefetch(&A[(size_t)(rowBase + a_r) * K + k0 + 32], 0, 0);
            __builtin_prefetch(&B[(size_t)(k0 + 32 + b_kk) * N + colBase + b_n8], 0, 0);
        }
        stage_wait();
        __syncthreads();

        // ---- fragments: all b128 LDS reads ----
        v16bf afrag[2];
#pragma unroll
        for (int tm = 0; tm < 2; ++tm)
            afrag[tm] = afrag_from_f32(&Asf[(waveM * 32 + tm * 16 + lo) * AS_STRIDE], hi);
        v16bf bfrag[4];
#pragma unroll
        for (int tn = 0; tn < 4; ++tn)
            bfrag[tn] = bfrag_from_bf16(&Bst[(waveN * 64 + tn * 16 + lo) * BS_STRIDE + hi * 8]);

#pragma unroll
        for (int tm = 0; tm < 2; ++tm)
#pragma unroll
            for (int tn = 0; tn < 4; ++tn)
                acc[tm][tn] = __builtin_amdgcn_wmma_f32_16x16x32_bf16(
                    false, afrag[tm], false, bfrag[tn],
                    (short)0, acc[tm][tn], false, false);
        __syncthreads();
    }

    // epilogue: C/D layout -> row = tileRow + hi*8 + r, col = tileCol + lo
#pragma unroll
    for (int tm = 0; tm < 2; ++tm)
#pragma unroll
        for (int tn = 0; tn < 4; ++tn) {
            int col = colBase + waveN * 64 + tn * 16 + lo;
            float bv = bias[col];
#pragma unroll
            for (int r = 0; r < 8; ++r) {
                int row = rowBase + waveM * 32 + tm * 16 + hi * 8 + r;
                float v = acc[tm][tn][r] + bv;
                if (relu) v = fmaxf(v, 0.0f);
                C[(size_t)row * N + col] = v;
            }
        }
}

// ---------------------------------------------------------------------------
// Flash attention, head_dim = 64. grid = (S/128, H, B), 256 threads.
// Each wave owns 16 query rows; K/V streamed in 32-key LDS chunks.
// ---------------------------------------------------------------------------
__global__ __launch_bounds__(256) void attn_flash_wmma(
    const float* __restrict__ Q, const float* __restrict__ K,
    const float* __restrict__ V, float* __restrict__ O,
    int S, int D)
{
    __shared__ __bf16 Ks[32 * KS_STRIDE];       // [key][hd]  (= B-layout rows) 4.5 KB
    __shared__ __bf16 Vst[TEL_HD * BS_STRIDE];  // [hd][key]  transposed        7 KB
    __shared__ __bf16 Ps[8][16 * BS_STRIDE];    // per-wave P staging           14 KB

    const int b = blockIdx.z, h = blockIdx.y;
    const int tid  = threadIdx.x;
    const int lane = tid & 31;
    const int wave = tid >> 5;
    const int lo = lane & 15;
    const int hi = lane >> 4;

    const size_t base = (size_t)b * S * D + (size_t)h * TEL_HD;
    const float* Qb = Q + base;
    const float* Kb = K + base;
    const float* Vb = V + base;
    float*       Ob = O + base;

    const int q0 = blockIdx.x * 128 + wave * 16;
    const float scale = 0.125f;   // 1/sqrt(64)

    // Q fragments in registers for the whole kernel: 4 runs of 8 contiguous K
    v16bf aq[2];
#pragma unroll
    for (int run = 0; run < 4; ++run) {
        const v4f* qp = (const v4f*)&Qb[(size_t)(q0 + lo) * D + run * 16 + hi * 8];
        v4f f0 = qp[0], f1 = qp[1];
#pragma unroll
        for (int e = 0; e < 4; ++e) {
            aq[run >> 1][(run & 1) * 8 + e]     = (__bf16)(f0[e] * scale);
            aq[run >> 1][(run & 1) * 8 + 4 + e] = (__bf16)(f1[e] * scale);
        }
    }

    const v8f vz = {0.f,0.f,0.f,0.f,0.f,0.f,0.f,0.f};
    v8f accO[4];
#pragma unroll
    for (int t = 0; t < 4; ++t) accO[t] = vz;
    float mrow[8], lrow[8];
#pragma unroll
    for (int r = 0; r < 8; ++r) { mrow[r] = -3.0e38f; lrow[r] = 0.0f; }

    const int st_key = tid >> 3;        // 0..31
    const int st_c8  = (tid & 7) * 8;   // 8 consecutive hd columns

    for (int j0 = 0; j0 < S; j0 += 32) {
        // ---- stage K chunk row-major [key][hd]: one v8bf store per thread ----
        {
            const v4f* gp = (const v4f*)&Kb[(size_t)(j0 + st_key) * D + st_c8];
            v4f f0 = gp[0], f1 = gp[1];
            v8bf kk;
#pragma unroll
            for (int j = 0; j < 4; ++j) { kk[j] = (__bf16)f0[j]; kk[4 + j] = (__bf16)f1[j]; }
            *(v8bf*)&Ks[st_key * KS_STRIDE + st_c8] = kk;
        }
        // ---- stage V chunk transposed [hd][key] ----
        {
            const v4f* gp = (const v4f*)&Vb[(size_t)(j0 + st_key) * D + st_c8];
            v4f f0 = gp[0], f1 = gp[1];
#pragma unroll
            for (int j = 0; j < 4; ++j) {
                Vst[(st_c8 + j) * BS_STRIDE + st_key]     = (__bf16)f0[j];
                Vst[(st_c8 + 4 + j) * BS_STRIDE + st_key] = (__bf16)f1[j];
            }
        }
        __syncthreads();

        // ---- S = Q @ K^T : 16 queries x 32 keys ----
        v8f accS[2] = {vz, vz};
#pragma unroll
        for (int tn = 0; tn < 2; ++tn)
#pragma unroll
            for (int c = 0; c < 2; ++c) {
                v16bf bk = bfrag_from_bf16(&Ks[(tn * 16 + lo) * KS_STRIDE + c * 32 + hi * 8]);
                accS[tn] = __builtin_amdgcn_wmma_f32_16x16x32_bf16(
                    false, aq[c], false, bk, (short)0, accS[tn], false, false);
            }

        // ---- online softmax; row M = hi*8 + r ----
#pragma unroll
        for (int r = 0; r < 8; ++r) {
            float s0 = accS[0][r], s1 = accS[1][r];
            float mx = fmaxf(s0, s1);
#pragma unroll
            for (int msk = 1; msk < 16; msk <<= 1)
                mx = fmaxf(mx, __shfl_xor(mx, msk, 32));
            float mnew  = fmaxf(mrow[r], mx);
            float alpha = __expf(mrow[r] - mnew);
            float p0 = __expf(s0 - mnew);
            float p1 = __expf(s1 - mnew);
            float ps = p0 + p1;
#pragma unroll
            for (int msk = 1; msk < 16; msk <<= 1)
                ps += __shfl_xor(ps, msk, 32);
            lrow[r] = lrow[r] * alpha + ps;
            mrow[r] = mnew;
#pragma unroll
            for (int t = 0; t < 4; ++t) accO[t][r] *= alpha;
            Ps[wave][(hi * 8 + r) * BS_STRIDE + lo]      = (__bf16)p0;
            Ps[wave][(hi * 8 + r) * BS_STRIDE + 16 + lo] = (__bf16)p1;
        }
        // per-wave LDS RAW: wave-local ordering is sufficient
        asm volatile("s_wait_dscnt 0" ::: "memory");

        // ---- P fragment (A-layout over 32 keys): two b128 reads ----
        v16bf ap;
        {
            const v8bf* p0 = (const v8bf*)&Ps[wave][lo * BS_STRIDE + hi * 8];
            const v8bf* p1 = (const v8bf*)&Ps[wave][lo * BS_STRIDE + 16 + hi * 8];
            ap = __builtin_shufflevector(p0[0], p1[0],
                                         0,1,2,3,4,5,6,7,8,9,10,11,12,13,14,15);
        }
#pragma unroll
        for (int t = 0; t < 4; ++t) {
            v16bf bv = bfrag_from_bf16(&Vst[(t * 16 + lo) * BS_STRIDE + hi * 8]);
            accO[t] = __builtin_amdgcn_wmma_f32_16x16x32_bf16(
                false, ap, false, bv, (short)0, accO[t], false, false);
        }
        __syncthreads();
    }

    // normalize and write ctx
#pragma unroll
    for (int t = 0; t < 4; ++t)
#pragma unroll
        for (int r = 0; r < 8; ++r) {
            int row = q0 + hi * 8 + r;
            Ob[(size_t)row * D + t * 16 + lo] = accO[t][r] / lrow[r];
        }
}

// ---------------------------------------------------------------------------
// out[row] = LayerNorm(A[row] + B[row]) * g + b     (D = 1024, 256 thr/row,
// one float4 per thread)
// ---------------------------------------------------------------------------
__global__ __launch_bounds__(256) void add_layernorm(
    const float* __restrict__ A, const float* __restrict__ Bv,
    const float* __restrict__ g, const float* __restrict__ bta,
    float* __restrict__ out, int D)
{
    __shared__ float red0[8], red1[8];
    const int row = blockIdx.x;
    const int tid = threadIdx.x;
    const int lane = tid & 31, wave = tid >> 5;

    v4f va = *(const v4f*)&A[(size_t)row * D + tid * 4];
    v4f vb = *(const v4f*)&Bv[(size_t)row * D + tid * 4];
    v4f v  = va + vb;
    float s = v[0] + v[1] + v[2] + v[3];
    float s2 = v[0]*v[0] + v[1]*v[1] + v[2]*v[2] + v[3]*v[3];
#pragma unroll
    for (int msk = 1; msk < 32; msk <<= 1) {
        s  += __shfl_xor(s,  msk, 32);
        s2 += __shfl_xor(s2, msk, 32);
    }
    if (lane == 0) { red0[wave] = s; red1[wave] = s2; }
    __syncthreads();
    float ts = 0.f, ts2 = 0.f;
#pragma unroll
    for (int w = 0; w < 8; ++w) { ts += red0[w]; ts2 += red1[w]; }
    float mu   = ts / (float)D;
    float var  = ts2 / (float)D - mu * mu;
    float rstd = rsqrtf(var + 1e-5f);
    v4f vg = *(const v4f*)&g[tid * 4];
    v4f vt = *(const v4f*)&bta[tid * 4];
    v4f o;
#pragma unroll
    for (int e = 0; e < 4; ++e) o[e] = (v[e] - mu) * rstd * vg[e] + vt[e];
    *(v4f*)&out[(size_t)row * D + tid * 4] = o;
}

// ---------------------------------------------------------------------------
extern "C" void kernel_launch(void* const* d_in, const int* in_sizes, int n_in,
                              void* d_out, int out_size, void* d_ws, size_t ws_size,
                              hipStream_t stream) {
    (void)in_sizes; (void)n_in; (void)out_size; (void)ws_size;
    const float* src   = (const float*)d_in[0];
    const float* Wq    = (const float*)d_in[1];
    const float* bq    = (const float*)d_in[2];
    const float* Wk    = (const float*)d_in[3];
    const float* bk    = (const float*)d_in[4];
    const float* Wv    = (const float*)d_in[5];
    const float* bv    = (const float*)d_in[6];
    const float* Wo    = (const float*)d_in[7];
    const float* bo    = (const float*)d_in[8];
    const float* W1    = (const float*)d_in[9];
    const float* b1    = (const float*)d_in[10];
    const float* W2    = (const float*)d_in[11];
    const float* b2    = (const float*)d_in[12];
    const float* ln1_g = (const float*)d_in[13];
    const float* ln1_b = (const float*)d_in[14];
    const float* ln2_g = (const float*)d_in[15];
    const float* ln2_b = (const float*)d_in[16];

    const int D = TEL_D, F = TEL_F, S = TEL_S, B = TEL_B;
    const int M = B * S;                       // 4096 token rows
    const size_t MD = (size_t)M * D;

    float* Q    = (float*)d_ws;
    float* Kbuf = Q    + MD;
    float* Vbuf = Kbuf + MD;
    float* ctx  = Vbuf + MD;
    float* tmp  = ctx  + MD;
    float* x    = tmp  + MD;
    float* hbuf = x    + MD;                   // M x F

    dim3 blk(256);
    dim3 gDD(D / 128, M / 128);
    dim3 gDF(F / 128, M / 128);

    // QKV projections
    gemm_bf16_wmma<<<gDD, blk, 0, stream>>>(src, Wq, bq, Q,    M, D, D, 0);
    gemm_bf16_wmma<<<gDD, blk, 0, stream>>>(src, Wk, bk, Kbuf, M, D, D, 0);
    gemm_bf16_wmma<<<gDD, blk, 0, stream>>>(src, Wv, bv, Vbuf, M, D, D, 0);
    // attention
    attn_flash_wmma<<<dim3(S / 128, 16, B), blk, 0, stream>>>(Q, Kbuf, Vbuf, ctx, S, D);
    // output projection + residual LN
    gemm_bf16_wmma<<<gDD, blk, 0, stream>>>(ctx, Wo, bo, tmp, M, D, D, 0);
    add_layernorm<<<M, blk, 0, stream>>>(src, tmp, ln1_g, ln1_b, x, D);
    // FFN
    gemm_bf16_wmma<<<gDF, blk, 0, stream>>>(x, W1, b1, hbuf, M, F, D, 1);
    gemm_bf16_wmma<<<gDD, blk, 0, stream>>>(hbuf, W2, b2, tmp, M, D, F, 0);
    add_layernorm<<<M, blk, 0, stream>>>(x, tmp, ln2_g, ln2_b, (float*)d_out, D);
}
